// MirrorSystem_24902220382482
// MI455X (gfx1250) — compile-verified, matrix operand order checked by reading, not run on hardware
//
#include <hip/hip_runtime.h>

// ---------------------------------------------------------------------------
// MirrorSystem fused VQ pipeline for gfx1250 (CDNA5, wave32, WMMA + TDM).
// B=8, S=4096, DIM=256, NSYM=1024.  M = B*S = 32768 rows.
// ---------------------------------------------------------------------------

#define M_TOTAL   32768
#define DIMC      256
#define KDIM      512          // 2*DIM
#define NSYM      1024
#define N_ELEM    8388608      // B*S*DIM, for variance
#define OUT_ZQ    16777216     // 32768*512

typedef __attribute__((ext_vector_type(16))) __bf16 v16bf;
typedef __attribute__((ext_vector_type(8)))  float  v8f;

// TDM descriptor group types (per CDNA5_HIP.md probe results)
typedef __attribute__((ext_vector_type(4))) unsigned int tdm_g0_t;
typedef __attribute__((ext_vector_type(8))) int          tdm_g1_t;
typedef __attribute__((ext_vector_type(4))) int          tdm_g2_t;
typedef __attribute__((ext_vector_type(8))) int          tdm_g4_t;

// ---- workspace layout (bytes) ---------------------------------------------
#define WS_SCAL     0            // float[8]: [0]=sum|z| [1]=sum|z|^2 [2]=energy acc [3]=energy_weight
#define WS_ZNSQ     256          // float[32768]
#define WS_ZNORM    131328       // bf16 [32768][512]
#define WS_CBB      33685760     // bf16 [1024][512]   codebook row-major
#define WS_CBT      34734336     // bf16 [512][1024]   codebook transposed
#define WS_ADJT     35782912     // bf16 [1024][1024]  adjacency transposed
#define WS_CNORM    37880064     // float[1024]

// ---- main-kernel LDS layout -----------------------------------------------
// TDM pads 4 DWORDs after every 256 DWORDs stored (pad_interval=7, pad_amount=3).
#define ZP   520                 // bf16 row: 512 + 8 (one pad group per row)
#define PP   1040                // f32 row: 1024 + 4*4 (four pad groups per row)
#define AP   1040                // bf16 row for probs (overlay on prev region)
#define DP   1032                // f32 row of d_total tile
#define SM_Z 0
#define SM_P (16*ZP*2)                       // 16640
#define SM_D (SM_P + 16*PP*4)                // 83200
#define KD_SMEM (SM_D + 16*DP*4)             // 149248 bytes

// ---------------------------------------------------------------------------
__device__ __forceinline__ float wave_sum(float v) {
#pragma unroll
  for (int m = 16; m >= 1; m >>= 1) v += __shfl_xor(v, m, 32);
  return v;
}
__device__ __forceinline__ float wave_max(float v) {
#pragma unroll
  for (int m = 16; m >= 1; m >>= 1) v = fmaxf(v, __shfl_xor(v, m, 32));
  return v;
}

// Pack 16 bf16 of one operand row into the 16x16x32 bf16 fragment layout:
// VGPR v in 0..3 holds K = 2v, 2v+1 ; VGPR v in 4..7 holds K = 16+2(v-4), +1.
// Caller already applied the +8 K-offset for lanes 16..31 and the k-step.
__device__ __forceinline__ v16bf frag_row(const __bf16* rowbase) {
  v16bf f;
#pragma unroll
  for (int v = 0; v < 4; ++v) {
    f[2*v]     = rowbase[2*v];
    f[2*v + 1] = rowbase[2*v + 1];
    f[2*v + 8] = rowbase[16 + 2*v];
    f[2*v + 9] = rowbase[16 + 2*v + 1];
  }
  return f;
}

// Same fragment, but source row is f32 in LDS with the TDM pad skew
// (+4 floats after every 256), converted to bf16 at build time.
__device__ __forceinline__ v16bf frag_row_f32(const float* row, int k0) {
  const float* p0 = row + k0 + ((k0 >> 8) << 2);
  const int k1 = k0 + 16;
  const float* p1 = row + k1 + ((k1 >> 8) << 2);
  v16bf f;
#pragma unroll
  for (int v = 0; v < 4; ++v) {
    f[2*v]     = (__bf16)p0[2*v];
    f[2*v + 1] = (__bf16)p0[2*v + 1];
    f[2*v + 8] = (__bf16)p1[2*v];
    f[2*v + 9] = (__bf16)p1[2*v + 1];
  }
  return f;
}

#if __has_builtin(__builtin_amdgcn_tensor_load_to_lds)
#define HAVE_TDM 1
// Build D# group 0: count=1, lds byte addr, 57b global addr, type=2.
__device__ __forceinline__ tdm_g0_t tdm_group0(unsigned lds_addr,
                                               unsigned long long gaddr) {
  tdm_g0_t g;
  g[0] = 1u;                                        // count=1, user mode
  g[1] = lds_addr;                                  // LDS address (bytes)
  g[2] = (unsigned)(gaddr & 0xFFFFFFFFull);         // global_addr[31:0]
  g[3] = (unsigned)((gaddr >> 32) & 0x01FFFFFFull)  // global_addr[56:32]
         | (2u << 30);                              // type=2 ("image")
  return g;
}
// Build D# group 1: 2D tile, data_size code, pad 4 DWORDs / 256 DWORDs.
__device__ __forceinline__ tdm_g1_t tdm_group1(unsigned data_size_code,
                                               unsigned tensor_d0,
                                               unsigned tensor_d1,
                                               unsigned tile_d0,
                                               unsigned tile_d1,
                                               unsigned long long stride0) {
  tdm_g1_t g;
  g[0] = (int)((data_size_code << 16) | (1u << 20)  // pad_enable
               | (7u << 22)                         // pad_interval: 256 DWORDs
               | (3u << 25));                       // pad_amount:   4 DWORDs
  g[1] = (int)((tensor_d0 & 0xFFFFu) << 16);        // tensor_dim0[15:0]
  g[2] = (int)(((tensor_d0 >> 16) & 0xFFFFu) | ((tensor_d1 & 0xFFFFu) << 16));
  g[3] = (int)(((tensor_d1 >> 16) & 0xFFFFu) | ((tile_d0 & 0xFFFFu) << 16));
  g[4] = (int)(tile_d1 & 0xFFFFu);                  // tile_dim2 = 0
  g[5] = (int)(stride0 & 0xFFFFFFFFull);            // tensor_dim0_stride lo
  g[6] = (int)((stride0 >> 32) & 0xFFFFu);          // stride hi | dim1_stride=0
  g[7] = 0;
  return g;
}
__device__ __forceinline__ void tdm_load(tdm_g0_t g0, tdm_g1_t g1) {
  const tdm_g2_t gz4 = {0, 0, 0, 0};
#if defined(__clang_major__) && (__clang_major__ >= 23)
  const tdm_g4_t gz8 = {0, 0, 0, 0, 0, 0, 0, 0};
  __builtin_amdgcn_tensor_load_to_lds(g0, g1, gz4, gz4, gz8, 0);
#else
  __builtin_amdgcn_tensor_load_to_lds(g0, g1, gz4, gz4, 0);
#endif
}
#else
#define HAVE_TDM 0
#endif

// ---------------------------------------------------------------------------
__global__ void k_zero(float* scal) {
  if (threadIdx.x < 8) scal[threadIdx.x] = 0.f;
}

// one wave per row: L2-normalize z_flat -> bf16, row ||z_norm||^2, and
// global sums of |z| and |z|^2 for the variance scalar.
__global__ void k_norm_var(const float* __restrict__ zr,
                           const float* __restrict__ zi,
                           __bf16* __restrict__ znorm,
                           float* __restrict__ znsq,
                           float* __restrict__ scal) {
  const int wave = threadIdx.x >> 5, lane = threadIdx.x & 31;
  const size_t row = (size_t)blockIdx.x * 8 + wave;
  const float* r  = zr + row * DIMC;
  const float* im = zi + row * DIMC;
  float zv[16];
  float ss = 0.f;
#pragma unroll
  for (int t = 0; t < 16; ++t) {
    const int d = lane + 32 * t;                 // 0..511 over z_flat
    const float x = (d < DIMC) ? r[d] : im[d - DIMC];
    zv[t] = x;
    ss += x * x;
  }
  ss = wave_sum(ss);
  const float inv = 1.f / (sqrtf(ss) + 1e-6f);
  __bf16* o = znorm + row * KDIM;
#pragma unroll
  for (int t = 0; t < 16; ++t) o[lane + 32 * t] = (__bf16)(zv[t] * inv);
  if (lane == 0) znsq[row] = ss * inv * inv;

  // |z|: element i pairs zv[t] (real, t<8) with zv[t+8] (imag)
  float ms = 0.f, m2 = 0.f;
#pragma unroll
  for (int t = 0; t < 8; ++t) {
    const float mg = sqrtf(zv[t] * zv[t] + zv[t + 8] * zv[t + 8]);
    ms += mg;
    m2 += mg * mg;
  }
  ms = wave_sum(ms);
  m2 = wave_sum(m2);
  if (lane == 0) {
    atomicAdd(&scal[0], ms);
    atomicAdd(&scal[1], m2);
  }
}

__global__ void k_scalar(float* scal) {
  if (threadIdx.x != 0 || blockIdx.x != 0) return;
  const double N = (double)N_ELEM;
  const double s = (double)scal[0], q = (double)scal[1];
  const double var = (q - s * s / N) / (N - 1.0);
  const float x = (float)(var / (1.0 + 1e-6));
  const float up = (x > 20.f) ? x : log1pf(expf(x));   // softplus
  scal[3] = 0.1f * up;                                  // lambda_energy * up
}

__global__ void k_cvt_cb(const float* __restrict__ cb,
                         __bf16* __restrict__ cbb,
                         __bf16* __restrict__ cbt) {
  const int i = blockIdx.x * 256 + threadIdx.x;         // < 1024*512
  const int n = i >> 9, d = i & 511;
  const float v = cb[i];
  cbb[i] = (__bf16)v;
  cbt[d * NSYM + n] = (__bf16)v;
}

__global__ void k_cnorm(const float* __restrict__ cb, float* __restrict__ cn) {
  const int wave = threadIdx.x >> 5, lane = threadIdx.x & 31;
  const int n = blockIdx.x * 8 + wave;
  const float* r = cb + (size_t)n * KDIM;
  float s = 0.f;
#pragma unroll
  for (int t = 0; t < 16; ++t) {
    const float v = r[lane + 32 * t];
    s += v * v;
  }
  s = wave_sum(s);
  if (lane == 0) cn[n] = s;
}

__global__ void k_cvt_adjT(const float* __restrict__ adj, __bf16* __restrict__ at) {
  const int i = blockIdx.x * 256 + threadIdx.x;         // < 1024*1024
  const int n = i >> 10, k = i & 1023;
  at[i] = (__bf16)adj[k * NSYM + n];                    // at[n][k] = adj[k][n]
}

// ---------------------------------------------------------------------------
// Fused main kernel: 16 rows per workgroup, 256 threads (8 waves).
// Wave w owns NSYM columns [w*128, w*128+128) for GEMM1/2 and output
// columns [w*64, w*64+64) for GEMM3.  Input tiles staged by the TDM.
// ---------------------------------------------------------------------------
__global__ void __launch_bounds__(256)
k_main(const float* __restrict__ prev,
       const __bf16* __restrict__ znorm,
       const float* __restrict__ znsq,
       const __bf16* __restrict__ cbb,
       const __bf16* __restrict__ cbt,
       const __bf16* __restrict__ adjt,
       const float* __restrict__ cnorm,
       float* __restrict__ scal,
       float* __restrict__ out) {
  extern __shared__ __align__(16) char smem[];
  __bf16* z_lds  = (__bf16*)(smem + SM_Z);    // [16][ZP]  z_norm tile (bf16)
  float*  p_lds  = (float*)(smem + SM_P);     // [16][PP]  prev tile (f32, skewed)
  __bf16* pr_lds = (__bf16*)(smem + SM_P);    // [16][AP]  probs overlay (bf16)
  float*  d_lds  = (float*)(smem + SM_D);     // [16][DP]  d_total tile (f32)
  __shared__ float s_znsq[16];
  __shared__ float red[8];

  const int tid = threadIdx.x;
  const int wave = tid >> 5, lane = tid & 31;
  const int half = lane >> 4, nl = lane & 15;
  const size_t r0 = (size_t)blockIdx.x * 16;
  const float ew = scal[3];

  if (tid < 16) s_znsq[tid] = znsq[r0 + tid];

#if HAVE_TDM
  if (wave == 0) {
    const unsigned lds_base = (unsigned)(unsigned long long)(void*)smem;
    // z_norm tile: 512 x 16 bf16, row stride 512 elements
    tdm_load(tdm_group0(lds_base + SM_Z,
                        (unsigned long long)(const void*)(znorm + r0 * KDIM)),
             tdm_group1(/*data_size=2B*/ 1u, KDIM, 16u, KDIM, 16u,
                        (unsigned long long)KDIM));
    // prev_sym_dist tile: 1024 x 16 f32, row stride 1024 elements
    tdm_load(tdm_group0(lds_base + SM_P,
                        (unsigned long long)(const void*)(prev + r0 * NSYM)),
             tdm_group1(/*data_size=4B*/ 2u, NSYM, 16u, NSYM, 16u,
                        (unsigned long long)NSYM));
    __builtin_amdgcn_s_wait_tensorcnt(0);
  }
#else
  for (int i = tid; i < 16 * KDIM; i += 256) {
    const int m = i >> 9, d = i & 511;
    z_lds[m * ZP + d] = znorm[(r0 + m) * KDIM + d];
  }
  for (int i = tid; i < 16 * NSYM; i += 256) {
    const int m = i >> 10, n = i & 1023;
    p_lds[m * PP + n + ((n >> 8) << 2)] = prev[(r0 + m) * NSYM + n];
  }
#endif
  __syncthreads();

  const v8f vzero = {0.f, 0.f, 0.f, 0.f, 0.f, 0.f, 0.f, 0.f};

  // ---- GEMM1: d_content dot = z_norm[16x512] @ codebook^T -----------------
  v8f acc1[8];
#pragma unroll
  for (int t = 0; t < 8; ++t) acc1[t] = vzero;
#pragma unroll 1
  for (int k = 0; k < KDIM; k += 32) {
    const v16bf a = frag_row(z_lds + nl * ZP + k + half * 8);
#pragma unroll
    for (int t = 0; t < 8; ++t) {
      const int n = wave * 128 + t * 16 + nl;
      const v16bf b = frag_row(cbb + (size_t)n * KDIM + k + half * 8);
      acc1[t] = __builtin_amdgcn_wmma_f32_16x16x32_bf16(
          false, a, false, b, (short)0, acc1[t], false, false);
    }
  }

  // ---- GEMM2: graph_bias = prev[16x1024] @ adjacency ----------------------
  v8f acc2[8];
#pragma unroll
  for (int t = 0; t < 8; ++t) acc2[t] = vzero;
#pragma unroll 1
  for (int k = 0; k < NSYM; k += 32) {
    const v16bf a = frag_row_f32(p_lds + nl * PP, k + half * 8);
#pragma unroll
    for (int t = 0; t < 8; ++t) {
      const int n = wave * 128 + t * 16 + nl;
      const __bf16* brow = adjt + (size_t)n * NSYM + k + half * 8;
      __builtin_prefetch(brow + 32, 0, 0);       // global_prefetch_b8 next k-step
      const v16bf b = frag_row(brow);
      acc2[t] = __builtin_amdgcn_wmma_f32_16x16x32_bf16(
          false, a, false, b, (short)0, acc2[t], false, false);
    }
  }

  // ---- combine: d_total, energy accumulation ------------------------------
  float esum = 0.f;
#pragma unroll
  for (int t = 0; t < 8; ++t) {
    const int n = wave * 128 + t * 16 + nl;
    const float cn = cnorm[n];
#pragma unroll
    for (int r = 0; r < 8; ++r) {
      const int m = half ? (r + 8) : r;
      const float dc =
          fmaxf(s_znsq[m] + cn - 2.f * acc1[t][r], 0.f) * (1.f / (float)KDIM);
      const float sig = 1.f / (1.f + expf(-acc2[t][r]));
      const float dt = dc - ew * sig;
      esum += dt;
      d_lds[m * DP + n] = dt;
    }
  }
  esum = wave_sum(esum);
  if (lane == 0) red[wave] = esum;
  __syncthreads();
  if (tid == 0) {
    float s = 0.f;
#pragma unroll
    for (int w = 0; w < 8; ++w) s += red[w];
    atomicAdd(&scal[2], s);
  }

  // ---- row softmax over NSYM (wave w owns rows 2w, 2w+1) ------------------
  // prev tile is dead; its LDS region is reused for bf16 probs.
#pragma unroll 1
  for (int rr = 0; rr < 2; ++rr) {
    const int m = wave * 2 + rr;
    float mx = -1e30f;
    for (int j = lane; j < NSYM; j += 32) {
      const float v = fminf(fmaxf(-d_lds[m * DP + j], -50.f), 50.f);
      mx = fmaxf(mx, v);
    }
    mx = wave_max(mx);
    float s = 0.f;
    for (int j = lane; j < NSYM; j += 32) {
      const float v = fminf(fmaxf(-d_lds[m * DP + j], -50.f), 50.f);
      const float e = expf(v - mx);
      d_lds[m * DP + j] = e;
      s += e;
    }
    s = wave_sum(s);
    const float inv = 1.f / s;
    for (int j = lane; j < NSYM; j += 32)
      pr_lds[m * AP + j] = (__bf16)(d_lds[m * DP + j] * inv);
  }
  __syncthreads();

  // ---- GEMM3: z_q = probs[16x1024] @ codebook[1024x512] -------------------
  v8f acc3[4];
#pragma unroll
  for (int t = 0; t < 4; ++t) acc3[t] = vzero;
#pragma unroll 1
  for (int k = 0; k < NSYM; k += 32) {
    const v16bf a = frag_row(pr_lds + nl * AP + k + half * 8);
#pragma unroll
    for (int t = 0; t < 4; ++t) {
      const int n = wave * 64 + t * 16 + nl;
      const v16bf b = frag_row(cbt + (size_t)n * NSYM + k + half * 8);
      acc3[t] = __builtin_amdgcn_wmma_f32_16x16x32_bf16(
          false, a, false, b, (short)0, acc3[t], false, false);
    }
  }
#pragma unroll
  for (int t = 0; t < 4; ++t) {
#pragma unroll
    for (int r = 0; r < 8; ++r) {
      const int m = half ? (r + 8) : r;
      const int n = wave * 64 + t * 16 + nl;
      out[(r0 + m) * KDIM + n] = acc3[t][r];
    }
  }
}

__global__ void k_energy(const float* scal, float* out) {
  if (threadIdx.x == 0 && blockIdx.x == 0)
    out[OUT_ZQ] = scal[2] / ((float)M_TOTAL * (float)NSYM);
}

// ---------------------------------------------------------------------------
extern "C" void kernel_launch(void* const* d_in, const int* in_sizes, int n_in,
                              void* d_out, int out_size, void* d_ws,
                              size_t ws_size, hipStream_t stream) {
  (void)in_sizes; (void)n_in; (void)out_size; (void)ws_size;
  const float* zr   = (const float*)d_in[0];
  const float* zi   = (const float*)d_in[1];
  const float* prev = (const float*)d_in[2];
  const float* cb   = (const float*)d_in[3];
  const float* adj  = (const float*)d_in[4];
  float* out = (float*)d_out;

  char* ws = (char*)d_ws;
  float*  scal  = (float*)(ws + WS_SCAL);
  float*  znsq  = (float*)(ws + WS_ZNSQ);
  __bf16* znorm = (__bf16*)(ws + WS_ZNORM);
  __bf16* cbb   = (__bf16*)(ws + WS_CBB);
  __bf16* cbt   = (__bf16*)(ws + WS_CBT);
  __bf16* adjt  = (__bf16*)(ws + WS_ADJT);
  float*  cnorm = (float*)(ws + WS_CNORM);

  (void)hipFuncSetAttribute((const void*)k_main,
                            hipFuncAttributeMaxDynamicSharedMemorySize,
                            KD_SMEM);

  k_zero<<<1, 64, 0, stream>>>(scal);
  k_norm_var<<<M_TOTAL / 8, 256, 0, stream>>>(zr, zi, znorm, znsq, scal);
  k_scalar<<<1, 64, 0, stream>>>(scal);
  k_cvt_cb<<<(NSYM * KDIM) / 256, 256, 0, stream>>>(cb, cbb, cbt);
  k_cnorm<<<NSYM / 8, 256, 0, stream>>>(cb, cnorm);
  k_cvt_adjT<<<(NSYM * NSYM) / 256, 256, 0, stream>>>(adj, adjt);
  k_main<<<M_TOTAL / 16, 256, KD_SMEM, stream>>>(prev, znorm, znsq, cbb, cbt,
                                                 adjt, cnorm, scal, out);
  k_energy<<<1, 64, 0, stream>>>(scal, out);
}
